// ContinuousIntegratedKoopmanOperator_52484500357780
// MI455X (gfx1250) — compile-verified, matrix operand order checked by reading, not run on hardware
//
#include <hip/hip_runtime.h>

// ---------------------------------------------------------------------------
// Problem constants (reference: B=8192, D=128, T=64, dt=0.01)
// ---------------------------------------------------------------------------
#define BB 8192
#define DD 128
#define TT 64
#define DT_STEP 0.01f
#define MAT_ELEMS (DD * DD) // 16384

typedef _Float16 v16h __attribute__((ext_vector_type(16)));
typedef _Float16 h8   __attribute__((ext_vector_type(8)));
typedef float    v8f  __attribute__((ext_vector_type(8)));

__device__ __forceinline__ v8f wmma_f16(v16h a, v16h b, v8f c) {
  // 8 args: (neg_a, A, neg_b, B, c_mod, C, reuse_a, reuse_b)
  return __builtin_amdgcn_wmma_f32_16x16x32_f16(false, a, false, b, (short)0, c,
                                                false, false);
}

// ---------------------------------------------------------------------------
// S = dt * W
// ---------------------------------------------------------------------------
__global__ __launch_bounds__(256) void scale_kernel(const float* __restrict__ W,
                                                    float* __restrict__ S) {
  int e = blockIdx.x * 256 + threadIdx.x;
  S[e] = DT_STEP * W[e];
}

// ---------------------------------------------------------------------------
// One-workgroup 128x128x128 fp32 GEMM: C = alpha * (A @ B), via split-f16 WMMA.
// 256 threads = 8 waves; wave w owns output row-tile w (rows 16w..16w+15).
// A fragment (16x32 f16): lanes 0-15 -> K=kb..kb+7 in elems 0..7, K=16+kb..
// in elems 8..15, with kb = (lane>=16 ? 8 : 0). B mirrors with N = lane&15.
// ---------------------------------------------------------------------------
__device__ __forceinline__ void device_gemm128(const float* __restrict__ A,
                                               const float* __restrict__ B,
                                               float* __restrict__ C,
                                               float alpha) {
  const int tid  = threadIdx.x;
  const int wave = tid >> 5;
  const int lane = tid & 31;
  const int lrow = lane & 15;
  const int khi  = lane >> 4;   // 0 or 1
  const int kb   = khi * 8;
  const int r    = wave * 16 + lrow;

  v16h ahi[4], alo[4];
#pragma unroll
  for (int kt = 0; kt < 4; ++kt) {
#pragma unroll
    for (int j = 0; j < 8; ++j) {
      float v0 = A[r * DD + kt * 32 + kb + j];
      _Float16 h0 = (_Float16)v0;
      ahi[kt][j] = h0;
      alo[kt][j] = (_Float16)(v0 - (float)h0);
      float v1 = A[r * DD + kt * 32 + 16 + kb + j];
      _Float16 h1 = (_Float16)v1;
      ahi[kt][j + 8] = h1;
      alo[kt][j + 8] = (_Float16)(v1 - (float)h1);
    }
  }

  for (int nt = 0; nt < 8; ++nt) {
    const int c = nt * 16 + lrow;
    v8f acc = {0.f, 0.f, 0.f, 0.f, 0.f, 0.f, 0.f, 0.f};
#pragma unroll
    for (int kt = 0; kt < 4; ++kt) {
      v16h bhi, blo;
#pragma unroll
      for (int j = 0; j < 8; ++j) {
        float v0 = B[(kt * 32 + kb + j) * DD + c];
        _Float16 h0 = (_Float16)v0;
        bhi[j] = h0;
        blo[j] = (_Float16)(v0 - (float)h0);
        float v1 = B[(kt * 32 + 16 + kb + j) * DD + c];
        _Float16 h1 = (_Float16)v1;
        bhi[j + 8] = h1;
        blo[j + 8] = (_Float16)(v1 - (float)h1);
      }
      acc = wmma_f16(ahi[kt], bhi, acc);
      acc = wmma_f16(ahi[kt], blo, acc);
      acc = wmma_f16(alo[kt], bhi, acc);
    }
    // C layout: VGPR j holds M=j (lanes 0-15) / M=j+8 (lanes 16-31), N=lane&15
#pragma unroll
    for (int j = 0; j < 8; ++j) {
      C[(wave * 16 + j + khi * 8) * DD + c] = alpha * acc[j];
    }
  }
}

__global__ __launch_bounds__(256) void gemm128_kernel(const float* __restrict__ A,
                                                      const float* __restrict__ B,
                                                      float* __restrict__ C,
                                                      float alpha) {
  device_gemm128(A, B, C, alpha);
}

// ---------------------------------------------------------------------------
// P[0] = I ; P[1] = I + S + S^2/2! + S^3/3! + S^4/4! + S^5/5!
// (||S|| ~ 0.02 -> truncation error ~1e-13, far below fp32)
// ---------------------------------------------------------------------------
__global__ __launch_bounds__(256) void combine_kernel(const float* __restrict__ S,
                                                      const float* __restrict__ T,
                                                      float* __restrict__ P) {
  int e = blockIdx.x * 256 + threadIdx.x;
  int r = e >> 7, c = e & 127;
  float id = (r == c) ? 1.0f : 0.0f;
  P[e] = id; // P^0 = I
  P[MAT_ELEMS + e] = id + S[e] + T[e] + T[MAT_ELEMS + e] + T[2 * MAT_ELEMS + e] +
                     T[3 * MAT_ELEMS + e];
}

// ---------------------------------------------------------------------------
// Doubling stage: block i computes P[m+1+i] = P[i+1] @ P[m]   (powers 0-indexed)
// ---------------------------------------------------------------------------
__global__ __launch_bounds__(256) void pow_stage_kernel(float* __restrict__ P, int m) {
  const int i = blockIdx.x;
  device_gemm128(P + (size_t)(i + 1) * MAT_ELEMS,
                 P + (size_t)m * MAT_ELEMS,
                 P + (size_t)(m + 1 + i) * MAT_ELEMS, 1.0f);
}

// ---------------------------------------------------------------------------
// Split P^t (fp32, [k][n]) into transposed f16 hi/lo tables PT[t][n][k]
// so B fragments become contiguous 16B loads per lane in the main kernel.
// 64*128*128 halves per table; 8 elements per thread.
// ---------------------------------------------------------------------------
__global__ __launch_bounds__(256) void split_transpose_kernel(
    const float* __restrict__ P, _Float16* __restrict__ PThi,
    _Float16* __restrict__ PTlo) {
  int gid = blockIdx.x * 256 + threadIdx.x;
  int e   = gid * 8;          // linear index into [t][n][k]
  int t   = e >> 14;
  int rem = e & 16383;
  int n   = rem >> 7;
  int k0  = rem & 127;
  const float* Pt = P + (size_t)t * MAT_ELEMS;
#pragma unroll
  for (int j = 0; j < 8; ++j) {
    float v = Pt[(k0 + j) * DD + n]; // transpose: PT[n][k] = P[k][n]
    _Float16 h = (_Float16)v;
    PThi[e + j] = h;
    PTlo[e + j] = (_Float16)(v - (float)h);
  }
}

// ---------------------------------------------------------------------------
// Main kernel: out[b, t, :] = x[b, :] @ P^t.
// Grid (B/128, T/16), 256 threads = 8 waves. Wave w owns rows
// blockIdx.x*128 + 16w; A fragments (split f16) loaded once from x, then
// 16 time steps x 8 N-tiles x 4 K-chunks x 3 WMMAs. Propagator tables are
// 4 MB total -> L2-resident broadcast to all WGPs; traffic is dominated by
// the 268 MB output stream (memory floor ~12 us at 23.3 TB/s).
// ---------------------------------------------------------------------------
__global__ __launch_bounds__(256) void koopman_out_kernel(
    const float* __restrict__ x, const _Float16* __restrict__ PThi,
    const _Float16* __restrict__ PTlo, float* __restrict__ out) {
  const int tid  = threadIdx.x;
  const int wave = tid >> 5;
  const int lane = tid & 31;
  const int lrow = lane & 15;
  const int khi  = lane >> 4;
  const int kb   = khi * 8;
  const int rowBase = blockIdx.x * 128 + wave * 16;
  const int r       = rowBase + lrow;

  // A fragments (hi/lo split) for this wave's 16 x 128 block of x.
  v16h ahi[4], alo[4];
#pragma unroll
  for (int kt = 0; kt < 4; ++kt) {
#pragma unroll
    for (int j = 0; j < 8; ++j) {
      float v0 = x[(size_t)r * DD + kt * 32 + kb + j];
      _Float16 h0 = (_Float16)v0;
      ahi[kt][j] = h0;
      alo[kt][j] = (_Float16)(v0 - (float)h0);
      float v1 = x[(size_t)r * DD + kt * 32 + 16 + kb + j];
      _Float16 h1 = (_Float16)v1;
      ahi[kt][j + 8] = h1;
      alo[kt][j + 8] = (_Float16)(v1 - (float)h1);
    }
  }

  const int t0 = blockIdx.y * 16;
  for (int t = t0; t < t0 + 16; ++t) {
    const _Float16* __restrict__ Bh = PThi + (size_t)t * MAT_ELEMS;
    const _Float16* __restrict__ Bl = PTlo + (size_t)t * MAT_ELEMS;
    for (int nt = 0; nt < 8; ++nt) {
      const int n = nt * 16 + lrow;
      v8f acc = {0.f, 0.f, 0.f, 0.f, 0.f, 0.f, 0.f, 0.f};
#pragma unroll
      for (int kt = 0; kt < 4; ++kt) {
        h8 b0 = *(const h8*)(Bh + (size_t)n * DD + kt * 32 + kb);
        h8 b1 = *(const h8*)(Bh + (size_t)n * DD + kt * 32 + 16 + kb);
        h8 l0 = *(const h8*)(Bl + (size_t)n * DD + kt * 32 + kb);
        h8 l1 = *(const h8*)(Bl + (size_t)n * DD + kt * 32 + 16 + kb);
        v16h bhi = __builtin_shufflevector(b0, b1, 0, 1, 2, 3, 4, 5, 6, 7, 8, 9,
                                           10, 11, 12, 13, 14, 15);
        v16h blo = __builtin_shufflevector(l0, l1, 0, 1, 2, 3, 4, 5, 6, 7, 8, 9,
                                           10, 11, 12, 13, 14, 15);
        acc = wmma_f16(ahi[kt], bhi, acc);
        acc = wmma_f16(ahi[kt], blo, acc);
        acc = wmma_f16(alo[kt], bhi, acc);
      }
#pragma unroll
      for (int j = 0; j < 8; ++j) {
        out[((size_t)(rowBase + j + khi * 8) * TT + t) * DD + n] = acc[j];
      }
    }
  }
}

// ---------------------------------------------------------------------------
// Host-side orchestration (all launches on `stream`; workspace layout:
//   floats:  S[16384] | P[64*16384] | Ttmp[4*16384]
//   halves:  PThi[64*16384] | PTlo[64*16384]
// total ~8.5 MB)
// ---------------------------------------------------------------------------
extern "C" void kernel_launch(void* const* d_in, const int* in_sizes, int n_in,
                              void* d_out, int out_size, void* d_ws,
                              size_t ws_size, hipStream_t stream) {
  (void)in_sizes; (void)n_in; (void)out_size; (void)ws_size;

  const float* x = (const float*)d_in[0];
  const float* W = (const float*)d_in[1];
  float* out = (float*)d_out;

  float* ws   = (float*)d_ws;
  float* S    = ws;                                  // 16384 floats
  float* P    = ws + MAT_ELEMS;                      // 64 * 16384 floats
  float* Ttmp = ws + (size_t)MAT_ELEMS * 65;         // 4  * 16384 floats
  _Float16* PThi = (_Float16*)(ws + (size_t)MAT_ELEMS * 69);
  _Float16* PTlo = PThi + (size_t)TT * MAT_ELEMS;

  // 1) S = dt * W
  scale_kernel<<<MAT_ELEMS / 256, 256, 0, stream>>>(W, S);

  // 2) Taylor terms: T2 = S*S/2, T3 = T2*S/3, T4 = T3*S/4, T5 = T4*S/5
  gemm128_kernel<<<1, 256, 0, stream>>>(S, S, Ttmp, 0.5f);
  gemm128_kernel<<<1, 256, 0, stream>>>(Ttmp, S, Ttmp + MAT_ELEMS, 1.0f / 3.0f);
  gemm128_kernel<<<1, 256, 0, stream>>>(Ttmp + MAT_ELEMS, S, Ttmp + 2 * MAT_ELEMS, 0.25f);
  gemm128_kernel<<<1, 256, 0, stream>>>(Ttmp + 2 * MAT_ELEMS, S, Ttmp + 3 * MAT_ELEMS, 0.2f);

  // 3) P[0] = I, P[1] = expm(S)
  combine_kernel<<<MAT_ELEMS / 256, 256, 0, stream>>>(S, Ttmp, P);

  // 4) Power table by doubling: after stage m, P[0..2m] valid (need up to 63)
  for (int m = 1; m < TT; m <<= 1) {
    int nblk = (m < (TT - 1 - m)) ? m : (TT - 1 - m);
    pow_stage_kernel<<<nblk, 256, 0, stream>>>(P, m);
  }

  // 5) Split + transpose table into f16 hi/lo: PT[t][n][k]
  split_transpose_kernel<<<(TT * MAT_ELEMS / 8) / 256, 256, 0, stream>>>(P, PThi, PTlo);

  // 6) Main batched GEMM: out[b, t, :] = x[b, :] @ P^t
  koopman_out_kernel<<<dim3(BB / 128, TT / 16), 256, 0, stream>>>(x, PThi, PTlo, out);
}
